// MultiHeadAttention_32152125178397
// MI455X (gfx1250) — matrix operand order repaired and checked by reading, not compiled, not run on hardware
//
#include <hip/hip_runtime.h>
#include <math.h>

#define B_  2
#define S_  4096
#define D_  512
#define H_  8
#define DK_ 64

typedef __attribute__((ext_vector_type(16))) __bf16 v16bf;
typedef __attribute__((ext_vector_type(8)))  float  v8f;

static __device__ __forceinline__ v8f wmma_bf16(v16bf a, v16bf b, v8f c) {
    return __builtin_amdgcn_wmma_f32_16x16x32_bf16(
        /*neg_a=*/false, a, /*neg_b=*/false, b,
        /*c_mod=*/(short)0, c, /*reuse_a=*/false, /*reuse_b=*/false);
}

// 16 contiguous bf16 -> v16bf (two b128 loads)
static __device__ __forceinline__ v16bf ld16bf(const __bf16* p) {
    union { uint4 u[2]; v16bf v; } x;
    const uint4* q = (const uint4*)p;
    x.u[0] = q[0]; x.u[1] = q[1];
    return x.v;
}

// two separate runs of 8 bf16 -> v16bf (A-fragment layout)
static __device__ __forceinline__ v16bf ld8x2bf(const __bf16* p0, const __bf16* p1) {
    union { uint4 u[2]; v16bf v; } x;
    x.u[0] = *(const uint4*)p0;
    x.u[1] = *(const uint4*)p1;
    return x.v;
}

union BF8 { __bf16 h[8]; uint4 u; };

// ---------------------------------------------------------------------------
// One-shot fp32 -> bf16 conversion (vectorized, 8 elems/thread).
// count must be divisible by 2048 (true for all uses here).
// ---------------------------------------------------------------------------
__global__ __launch_bounds__(256) void cvt_f32_to_bf16(
    const float* __restrict__ in, __bf16* __restrict__ out)
{
    const size_t i = ((size_t)blockIdx.x * 256 + threadIdx.x) * 8;
    const float4 a = *(const float4*)(in + i);
    const float4 b = *(const float4*)(in + i + 4);
    BF8 r;
    r.h[0] = (__bf16)a.x; r.h[1] = (__bf16)a.y; r.h[2] = (__bf16)a.z; r.h[3] = (__bf16)a.w;
    r.h[4] = (__bf16)b.x; r.h[5] = (__bf16)b.y; r.h[6] = (__bf16)b.z; r.h[7] = (__bf16)b.w;
    *(uint4*)(out + i) = r.u;
}

// ---------------------------------------------------------------------------
// Projection: Y = X @ W^T, all-bf16 operands. Wave tile: 32 rows x 64 cols
// (strip == one head). 8 WMMAs per k-step from 12 b128 loads.
// transposed==0 : store bf16 [B,H,S,DK]   (Q, K)
// transposed==1 : store bf16 [B,H,DK,S]   (V^T -> contiguous P.V B-frags)
// ---------------------------------------------------------------------------
__global__ __launch_bounds__(256) void proj_kernel(
    const __bf16* __restrict__ X, const __bf16* __restrict__ W,
    __bf16* __restrict__ out, int transposed)
{
    const int lane = threadIdx.x & 31;
    const int wv   = threadIdx.x >> 5;
    const int gw   = blockIdx.x * 8 + wv;       // 0..2047
    const int rowTile = gw >> 3;                // 0..255 (32-row tiles)
    const int strip   = gw & 7;                 // head
    const int hi = lane >> 4;
    const int ln = lane & 15;

    const __bf16* ar0 = X + (size_t)(rowTile * 32 + ln) * D_;
    const __bf16* ar1 = ar0 + (size_t)16 * D_;

    v8f acc[2][4];
#pragma unroll
    for (int g = 0; g < 2; ++g)
#pragma unroll
        for (int t = 0; t < 4; ++t)
#pragma unroll
            for (int v = 0; v < 8; ++v) acc[g][t][v] = 0.0f;

    for (int k0 = 0; k0 < D_; k0 += 32) {
        const v16bf a0 = ld8x2bf(ar0 + k0 + 8 * hi, ar0 + k0 + 8 * hi + 16);
        const v16bf a1 = ld8x2bf(ar1 + k0 + 8 * hi, ar1 + k0 + 8 * hi + 16);
#pragma unroll
        for (int t = 0; t < 4; ++t) {
            const v16bf b = ld8x2bf(W + (size_t)(strip * 64 + 16 * t + ln) * D_ + k0 + 8 * hi, W + (size_t)(strip * 64 + 16 * t + ln) * D_ + k0 + 8 * hi + 16);
            acc[0][t] = wmma_bf16(a0, b, acc[0][t]);
            acc[1][t] = wmma_bf16(a1, b, acc[1][t]);
        }
    }

    const int m0 = rowTile * 32;
    const int b_ = m0 >> 12;          // batch (uniform: 32 | 4096)
    const int sb = m0 & (S_ - 1);
#pragma unroll
    for (int g = 0; g < 2; ++g)
#pragma unroll
        for (int t = 0; t < 4; ++t) {
            const int dk = 16 * t + ln;
            if (transposed) {
                BF8 r;
#pragma unroll
                for (int v = 0; v < 8; ++v) r.h[v] = (__bf16)acc[g][t][v];
                const int s0 = sb + 16 * g + 8 * hi;   // 8 consecutive s values
                *(uint4*)(out + ((size_t)((b_ * H_ + strip) * DK_ + dk)) * S_ + s0) = r.u;
            } else {
#pragma unroll
                for (int v = 0; v < 8; ++v) {
                    const int s = sb + 16 * g + v + 8 * hi;
                    out[((size_t)((b_ * H_ + strip) * S_ + s)) * DK_ + dk] = (__bf16)acc[g][t][v];
                }
            }
        }
}

// ---------------------------------------------------------------------------
// Flash attention (causal). One wave per 32-row query tile per (b,h):
// K/V fragment loads amortized over two 16-row groups -> 16 WMMA per 32 keys.
// ---------------------------------------------------------------------------
__global__ __launch_bounds__(256) void attn_kernel(
    const __bf16* __restrict__ Qh, const __bf16* __restrict__ Kh,
    const __bf16* __restrict__ Vt, __bf16* __restrict__ Ctx)
{
    __shared__ __align__(16) __bf16 ldsP[8][32][40];   // per-wave 32x32 P tile, padded

    const int lane = threadIdx.x & 31;
    const int wv   = threadIdx.x >> 5;
    const int gw   = blockIdx.x * 8 + wv;      // 0..2047
    const int qt   = gw & 127;
    const int bh   = gw >> 7;                  // 0..15
    const int q0   = qt * 32;
    const int hi = lane >> 4;
    const int ln = lane & 15;

    const __bf16* Qb = Qh + (size_t)bh * S_ * DK_;
    const __bf16* Kb = Kh + (size_t)bh * S_ * DK_;
    const __bf16* Vb = Vt + (size_t)bh * DK_ * S_;

    // Q fragments: 2 row-groups x 2 d-chunks
    v16bf aq[2][2];
#pragma unroll
    for (int g = 0; g < 2; ++g) {
        const __bf16* qr = Qb + (size_t)(q0 + 16 * g + ln) * DK_;
        aq[g][0] = ld8x2bf(qr + 0  + 16 * hi, qr + 0  + 16 * hi + 8);
        aq[g][1] = ld8x2bf(qr + 32 + 16 * hi, qr + 32 + 16 * hi + 8);
    }

    v8f acc[2][4];
    float mrow[2][8], lrow[2][8];
#pragma unroll
    for (int g = 0; g < 2; ++g) {
#pragma unroll
        for (int t = 0; t < 4; ++t)
#pragma unroll
            for (int v = 0; v < 8; ++v) acc[g][t][v] = 0.0f;
#pragma unroll
        for (int v = 0; v < 8; ++v) { mrow[g][v] = -__builtin_inff(); lrow[g][v] = 0.0f; }
    }

    for (int kb = 0; kb < q0 + 32; kb += 32) {
        // ---- K fragments for this 32-key block (shared by both row-groups) ----
        const __bf16* kr0 = Kb + (size_t)(kb + ln) * DK_;
        const __bf16* kr1 = Kb + (size_t)(kb + 16 + ln) * DK_;
        v16bf bk[2][2];
        bk[0][0] = ld16bf(kr0 + 0  + 16 * hi);
        bk[0][1] = ld16bf(kr0 + 32 + 16 * hi);
        bk[1][0] = ld16bf(kr1 + 0  + 16 * hi);
        bk[1][1] = ld16bf(kr1 + 32 + 16 * hi);

        // prefetch next key block (stays inside workspace; speculative-safe)
        __builtin_prefetch(kr0 + 32 * DK_, 0, 1);
        __builtin_prefetch(kr1 + 32 * DK_, 0, 1);

        float alpha[2][8];
#pragma unroll
        for (int g = 0; g < 2; ++g) {
            // ---- scores: two 16x16 tiles ----
            v8f s0, s1;
#pragma unroll
            for (int v = 0; v < 8; ++v) { s0[v] = 0.0f; s1[v] = 0.0f; }
            s0 = wmma_bf16(aq[g][0], bk[0][0], s0);
            s0 = wmma_bf16(aq[g][1], bk[0][1], s0);
            s1 = wmma_bf16(aq[g][0], bk[1][0], s1);
            s1 = wmma_bf16(aq[g][1], bk[1][1], s1);

            // ---- scale + causal mask + online softmax (fp32) ----
            const int c0 = kb + ln;
            const int c1 = kb + 16 + ln;
#pragma unroll
            for (int v = 0; v < 8; ++v) {
                const int rowg = q0 + 16 * g + v + 8 * hi;
                float x0 = s0[v] * 0.125f;               // 1/sqrt(64)
                float x1 = s1[v] * 0.125f;
                x0 = (rowg >= c0) ? x0 : -__builtin_inff();
                x1 = (rowg >= c1) ? x1 : -__builtin_inff();
                s0[v] = x0; s1[v] = x1;

                float mx = fmaxf(x0, x1);
                mx = fmaxf(mx, __shfl_xor(mx, 1, 32));
                mx = fmaxf(mx, __shfl_xor(mx, 2, 32));
                mx = fmaxf(mx, __shfl_xor(mx, 4, 32));
                mx = fmaxf(mx, __shfl_xor(mx, 8, 32));
                const float mn = fmaxf(mrow[g][v], mx);
                alpha[g][v] = __expf(mrow[g][v] - mn);
                mrow[g][v]  = mn;
            }
#pragma unroll
            for (int v = 0; v < 8; ++v) {
                const float p0 = __expf(s0[v] - mrow[g][v]);
                const float p1 = __expf(s1[v] - mrow[g][v]);
                float rs = p0 + p1;
                rs += __shfl_xor(rs, 1, 32);
                rs += __shfl_xor(rs, 2, 32);
                rs += __shfl_xor(rs, 4, 32);
                rs += __shfl_xor(rs, 8, 32);
                lrow[g][v] = lrow[g][v] * alpha[g][v] + rs;

                const int M = 16 * g + v + 8 * hi;
                ldsP[wv][M][ln]      = (__bf16)p0;       // D-layout -> LDS
                ldsP[wv][M][16 + ln] = (__bf16)p1;
            }
        }

        // ---- rescale running context ----
#pragma unroll
        for (int g = 0; g < 2; ++g)
#pragma unroll
            for (int t = 0; t < 4; ++t)
#pragma unroll
                for (int v = 0; v < 8; ++v) acc[g][t][v] *= alpha[g][v];

        // ---- P back in A-fragment layout (same-wave LDS ops are in-order) ----
        v16bf pa[2];
#pragma unroll
        for (int g = 0; g < 2; ++g)
            pa[g] = ld8x2bf(&ldsP[wv][16 * g + ln][8 * hi],
                            &ldsP[wv][16 * g + ln][8 * hi + 16]);

        // ---- context accumulation: each V fragment feeds 2 WMMAs ----
#pragma unroll
        for (int t = 0; t < 4; ++t) {
            const v16bf bv = ld8x2bf(Vb + (size_t)(16 * t + ln) * S_ + kb + 8 * hi, Vb + (size_t)(16 * t + ln) * S_ + kb + 8 * hi + 16);
            acc[0][t] = wmma_bf16(pa[0], bv, acc[0][t]);
            acc[1][t] = wmma_bf16(pa[1], bv, acc[1][t]);
        }
    }

    // ---- epilogue: normalize + store ctx bf16 in [B*S, 512] ----
    const int b_ = bh >> 3, h = bh & 7;
#pragma unroll
    for (int g = 0; g < 2; ++g)
#pragma unroll
        for (int t = 0; t < 4; ++t)
#pragma unroll
            for (int v = 0; v < 8; ++v) {
                const int rowg = q0 + 16 * g + v + 8 * hi;
                const size_t idx = ((size_t)(b_ * S_ + rowg)) * D_ + h * DK_ + 16 * t + ln;
                Ctx[idx] = (__bf16)(acc[g][t][v] / lrow[g][v]);
            }
}

// ---------------------------------------------------------------------------
// Output projection: Out = Ctx @ Wo^T + bo (bf16 operands -> f32 out)
// ---------------------------------------------------------------------------
__global__ __launch_bounds__(256) void outproj_kernel(
    const __bf16* __restrict__ Ctx, const __bf16* __restrict__ Wo,
    const float* __restrict__ bo, float* __restrict__ Out)
{
    const int lane = threadIdx.x & 31;
    const int wv   = threadIdx.x >> 5;
    const int gw   = blockIdx.x * 8 + wv;
    const int rowTile = gw >> 3;                // 32-row tiles
    const int strip   = gw & 7;
    const int hi = lane >> 4;
    const int ln = lane & 15;

    const __bf16* ar0 = Ctx + (size_t)(rowTile * 32 + ln) * D_;
    const __bf16* ar1 = ar0 + (size_t)16 * D_;

    v8f acc[2][4];
#pragma unroll
    for (int g = 0; g < 2; ++g)
#pragma unroll
        for (int t = 0; t < 4; ++t)
#pragma unroll
            for (int v = 0; v < 8; ++v) acc[g][t][v] = 0.0f;

    for (int k0 = 0; k0 < D_; k0 += 32) {
        const v16bf a0 = ld8x2bf(ar0 + k0 + 8 * hi, ar0 + k0 + 8 * hi + 16);
        const v16bf a1 = ld8x2bf(ar1 + k0 + 8 * hi, ar1 + k0 + 8 * hi + 16);
#pragma unroll
        for (int t = 0; t < 4; ++t) {
            const v16bf b = ld8x2bf(Wo + (size_t)(strip * 64 + 16 * t + ln) * D_ + k0 + 8 * hi, Wo + (size_t)(strip * 64 + 16 * t + ln) * D_ + k0 + 8 * hi + 16);
            acc[0][t] = wmma_bf16(a0, b, acc[0][t]);
            acc[1][t] = wmma_bf16(a1, b, acc[1][t]);
        }
    }

    const int m0 = rowTile * 32;
#pragma unroll
    for (int t = 0; t < 4; ++t) {
        const int n = strip * 64 + 16 * t + ln;
        const float bias = bo[n];
#pragma unroll
        for (int g = 0; g < 2; ++g)
#pragma unroll
            for (int v = 0; v < 8; ++v) {
                const int m = m0 + 16 * g + v + 8 * hi;
                Out[(size_t)m * D_ + n] = acc[g][t][v] + bias;
            }
    }
}

// ---------------------------------------------------------------------------
extern "C" void kernel_launch(void* const* d_in, const int* in_sizes, int n_in,
                              void* d_out, int out_size, void* d_ws, size_t ws_size,
                              hipStream_t stream) {
    const float* q   = (const float*)d_in[0];
    const float* k   = (const float*)d_in[1];
    const float* v   = (const float*)d_in[2];
    const float* W_q = (const float*)d_in[3];
    const float* W_k = (const float*)d_in[4];
    const float* W_v = (const float*)d_in[5];
    const float* W_o = (const float*)d_in[6];
    const float* b_o = (const float*)d_in[7];
    float* out = (float*)d_out;

    const size_t nX = (size_t)B_ * S_ * D_;     // 4,194,304
    const size_t nW = (size_t)D_ * D_;          // 262,144

    __bf16* Qh  = (__bf16*)d_ws;                // [B,H,S,DK]
    __bf16* Kh  = Qh  + nX;                     // [B,H,S,DK]
    __bf16* Vt  = Kh  + nX;                     // [B,H,DK,S]
    __bf16* Ctx = Vt  + nX;                     // [B*S, 512]
    __bf16* Xb  = Ctx + nX;                     // staging for current input
    __bf16* Wqb = Xb  + nX;
    __bf16* Wkb = Wqb + nW;
    __bf16* Wvb = Wkb + nW;
    __bf16* Wob = Wvb + nW;

    dim3 blk(256);
    const int xBlocks = (int)(nX / 2048);       // 2048
    const int wBlocks = (int)(nW / 2048);       // 128

    // one-shot weight conversions
    cvt_f32_to_bf16<<<wBlocks, blk, 0, stream>>>(W_q, Wqb);
    cvt_f32_to_bf16<<<wBlocks, blk, 0, stream>>>(W_k, Wkb);
    cvt_f32_to_bf16<<<wBlocks, blk, 0, stream>>>(W_v, Wvb);
    cvt_f32_to_bf16<<<wBlocks, blk, 0, stream>>>(W_o, Wob);

    // projections (input converted once each, then pure-bf16 GEMM)
    cvt_f32_to_bf16<<<xBlocks, blk, 0, stream>>>(q, Xb);
    proj_kernel<<<256, blk, 0, stream>>>(Xb, Wqb, Qh, 0);
    cvt_f32_to_bf16<<<xBlocks, blk, 0, stream>>>(k, Xb);
    proj_kernel<<<256, blk, 0, stream>>>(Xb, Wkb, Kh, 0);
    cvt_f32_to_bf16<<<xBlocks, blk, 0, stream>>>(v, Xb);
    proj_kernel<<<256, blk, 0, stream>>>(Xb, Wvb, Vt, 1);   // V stored transposed

    attn_kernel<<<256, blk, 0, stream>>>(Qh, Kh, Vt, Ctx);
    outproj_kernel<<<256, blk, 0, stream>>>(Ctx, Wob, b_o, out);
}
